// RWKV_48996986913351
// MI455X (gfx1250) — matrix operand order repaired and checked by reading, not compiled, not run on hardware
//
#include <hip/hip_runtime.h>
#include <hip/hip_bf16.h>
#include <math.h>

#define BQ 8
#define SQ 4096
#define SQ_SHIFT 12
#define DIN 256
#define HQ 512
#define LQ 2
#define MQ (BQ * SQ)

typedef __attribute__((ext_vector_type(16))) __bf16 bf16x16;
typedef __attribute__((ext_vector_type(8)))  float  f32x8;
typedef __attribute__((ext_vector_type(4)))  int    v4i;

struct alignas(16) V4 { unsigned x, y, z, w; };
union alignas(32) Frag { bf16x16 v; V4 q[2]; };

#if defined(__HIP_DEVICE_COMPILE__) && \
    __has_builtin(__builtin_amdgcn_global_load_async_to_lds_b128) && \
    __has_builtin(__builtin_amdgcn_s_wait_asynccnt)
#define USE_ASYNC_LDS 1
#else
#define USE_ASYNC_LDS 0
#endif

__device__ __forceinline__ unsigned short f2bf(float f) {
  union { float f; unsigned u; } c; c.f = f;
  unsigned u = c.u;
  unsigned r = (u + 0x7FFFu + ((u >> 16) & 1u)) >> 16;  // round-to-nearest-even
  return (unsigned short)r;
}

__device__ __forceinline__ float sigmoidf_dev(float x) {
  return 1.0f / (1.0f + __expf(-x));
}

#if USE_ASYNC_LDS
// global (as1) -> LDS (as3) 16-byte async copy, tracked by ASYNCcnt
__device__ __forceinline__ void async_copy16(const unsigned short* g,
                                             unsigned short* l) {
  __builtin_amdgcn_global_load_async_to_lds_b128(
      (__attribute__((address_space(1))) v4i*)(void*)g,
      (__attribute__((address_space(3))) v4i*)(void*)l, 0, 0);
}
__device__ __forceinline__ void wait_async_lds() {
  __builtin_amdgcn_s_wait_asynccnt(0);
}
#else
__device__ __forceinline__ void wait_async_lds() {}
#endif

// ---------------------------------------------------------------- elementwise
__global__ void cvt_bf16_kernel(const float* __restrict__ in,
                                unsigned short* __restrict__ out, int n) {
  for (int i = blockIdx.x * blockDim.x + threadIdx.x; i < n;
       i += gridDim.x * blockDim.x)
    out[i] = f2bf(in[i]);
}

// out[o,k] = bf16( W[o,k] * (tm ? tm[k] : 1) )
__global__ void prep_weight_kernel(const float* __restrict__ w,
                                   const float* __restrict__ tm,
                                   unsigned short* __restrict__ out,
                                   int N, int K) {
  int n = N * K;
  for (int i = blockIdx.x * blockDim.x + threadIdx.x; i < n;
       i += gridDim.x * blockDim.x) {
    int h = i % K;
    float f = w[i];
    if (tm) f *= tm[h];
    out[i] = f2bf(f);
  }
}

// ---------------------------------------------------------------- reductions
// mean over sequence: mean[b,c] = (1/S) sum_t h[b,t,c]
__global__ void seq_mean_kernel(const float* __restrict__ h,
                                float* __restrict__ mean) {
  int idx = blockIdx.x * blockDim.x + threadIdx.x;
  if (idx >= BQ * HQ) return;
  int b = idx / HQ, c = idx % HQ;
  const float* p = h + (size_t)b * SQ * HQ + c;
  float s = 0.f;
  for (int t = 0; t < SQ; ++t) s += p[(size_t)t * HQ];
  mean[idx] = s * (1.0f / SQ);
}

// bias[b,o] = sum_h W[o,h] * (1 - tm[h]) * mean[b,h]   (one wave per (b,o))
__global__ void mix_bias_kernel(const float* __restrict__ W,
                                const float* __restrict__ tm,
                                const float* __restrict__ mean,
                                float* __restrict__ bias) {
  int gw   = (blockIdx.x * blockDim.x + threadIdx.x) >> 5;
  int lane = threadIdx.x & 31;
  if (gw >= BQ * HQ) return;
  int b = gw / HQ, o = gw % HQ;
  const float* wr = W + (size_t)o * HQ;
  const float* mr = mean + (size_t)b * HQ;
  float s = 0.f;
  for (int h = lane; h < HQ; h += 32) s += wr[h] * (1.0f - tm[h]) * mr[h];
  for (int off = 16; off; off >>= 1) s += __shfl_down(s, off);
  if (lane == 0) bias[gw] = s;
}

// ---------------------------------------------------------------- wkv scan
// one lane per (b,channel); fuses sigmoid(k)*v, the scan, and r*wkv -> bf16
__global__ void wkv_scan_kernel(const float* __restrict__ k,
                                const float* __restrict__ v,
                                const float* __restrict__ r,
                                unsigned short* __restrict__ aOut) {
  int idx = blockIdx.x * blockDim.x + threadIdx.x;
  if (idx >= BQ * HQ) return;
  int b = idx / HQ, c = idx % HQ;
  size_t base = (size_t)b * SQ * HQ + c;
  float alpha = 0.f;
  for (int t = 0; t < SQ; ++t) {
    size_t off = base + (size_t)t * HQ;
    if (t + 8 < SQ) {  // pull future timesteps toward L2/L0 (global_prefetch_b8)
      size_t poff = off + (size_t)8 * HQ;
      __builtin_prefetch(&k[poff], 0, 1);
      __builtin_prefetch(&v[poff], 0, 1);
      __builtin_prefetch(&r[poff], 0, 1);
    }
    float s   = sigmoidf_dev(k[off]) * v[off];
    float wkv = s + alpha;
    aOut[off] = f2bf(r[off] * wkv);
    alpha = s + 0.9f * alpha;
  }
}

// ---------------------------------------------------------------- resid + LN
__device__ __forceinline__ float block_reduce_256(float v) {
  __shared__ float sh[8];
  int lane = threadIdx.x & 31, w = threadIdx.x >> 5;
  for (int o = 16; o; o >>= 1) v += __shfl_down(v, o);
  if (lane == 0) sh[w] = v;
  __syncthreads();
  float t = (threadIdx.x < 8) ? sh[threadIdx.x] : 0.f;
  if (w == 0) {
    for (int o = 4; o; o >>= 1) t += __shfl_down(t, o);
    if (lane == 0) sh[0] = t;
  }
  __syncthreads();
  float res = sh[0];
  __syncthreads();
  return res;
}

// y = LN(h+o, ln1); ff = relu(LN(y, ln2)); hnew = y + ff  (one block per row)
__global__ void resid_ln_kernel(const float* __restrict__ h,
                                const float* __restrict__ o,
                                const float* __restrict__ ln1w,
                                const float* __restrict__ ln1b,
                                const float* __restrict__ ln2w,
                                const float* __restrict__ ln2b,
                                float* __restrict__ hout,
                                unsigned short* __restrict__ hbf) {
  size_t base = (size_t)blockIdx.x * HQ;
  int t0 = threadIdx.x, t1 = threadIdx.x + 256;
  float a0 = h[base + t0] + o[base + t0];
  float a1 = h[base + t1] + o[base + t1];
  const float inv = 1.0f / HQ;
  float mu  = block_reduce_256(a0 + a1) * inv;
  float d0 = a0 - mu, d1 = a1 - mu;
  float var = block_reduce_256(d0 * d0 + d1 * d1) * inv;
  float rs  = rsqrtf(var + 1e-5f);
  float y0 = d0 * rs * ln1w[t0] + ln1b[t0];
  float y1 = d1 * rs * ln1w[t1] + ln1b[t1];
  float mu2  = block_reduce_256(y0 + y1) * inv;
  float e0 = y0 - mu2, e1 = y1 - mu2;
  float var2 = block_reduce_256(e0 * e0 + e1 * e1) * inv;
  float rs2  = rsqrtf(var2 + 1e-5f);
  float f0 = fmaxf(e0 * rs2 * ln2w[t0] + ln2b[t0], 0.f);
  float f1 = fmaxf(e1 * rs2 * ln2w[t1] + ln2b[t1], 0.f);
  float n0 = y0 + f0, n1 = y1 + f1;
  hout[base + t0] = n0;
  hout[base + t1] = n1;
  hbf[base + t0] = f2bf(n0);
  hbf[base + t1] = f2bf(n1);
}

// ---------------------------------------------------------------- WMMA GEMM
// C[M,N] = act( A[M,K](bf16) @ W[N,K]^T(bf16) + bias ), f32 accumulate.
// 128x128 tile, BK=32, double-buffered LDS filled by GLOBAL_LOAD_ASYNC_TO_LDS,
// 8 waves, 32x64 per wave, 8 WMMAs per K-step.
#define BM 128
#define BN 128
#define BK 32
#define LDT 40  // BK + 8 ushorts of padding -> conflict-free b128 frag reads

__launch_bounds__(256)
__global__ void gemm_bf16_wmma_kernel(const unsigned short* __restrict__ A,
                                      const unsigned short* __restrict__ W,
                                      float* __restrict__ C,
                                      unsigned short* __restrict__ Cbf,
                                      const float* __restrict__ bias,
                                      int N, int K,
                                      int act, int biasMode) {
  __shared__ unsigned short As[2][BM * LDT];
  __shared__ unsigned short Bs[2][BN * LDT];

  const int tid   = threadIdx.x;
  const int lane  = tid & 31;
  const int wid   = tid >> 5;
  const int waveM = wid >> 1;  // 0..3
  const int waveN = wid & 1;   // 0..1
  const size_t rowBase = (size_t)blockIdx.y * BM;
  const size_t colBase = (size_t)blockIdx.x * BN;

  f32x8 acc[2][4];
#pragma unroll
  for (int mi = 0; mi < 2; ++mi)
#pragma unroll
    for (int ni = 0; ni < 4; ++ni) acc[mi][ni] = (f32x8)0.0f;

  const int nStages = K / BK;

  // issue one stage's A+B tile copy into LDS buffer `buf`
  auto issue_tile = [&](int stage, int buf) {
    const size_t k0 = (size_t)stage * BK;
#pragma unroll
    for (int i = 0; i < 2; ++i) {
      int idx = tid + i * 256;  // 512 b128 segments per tile
      int row = idx >> 2;       // 4 segments per 32-col row
      int c16 = idx & 3;
      const unsigned short* ga = A + (rowBase + row) * K + k0 + c16 * 8;
      const unsigned short* gb = W + (colBase + row) * K + k0 + c16 * 8;
      unsigned short* la = &As[buf][row * LDT + c16 * 8];
      unsigned short* lb = &Bs[buf][row * LDT + c16 * 8];
#if USE_ASYNC_LDS
      async_copy16(ga, la);
      async_copy16(gb, lb);
#else
      V4 ta = *(const V4*)ga;
      V4 tb = *(const V4*)gb;
      *(V4*)la = ta;
      *(V4*)lb = tb;
#endif
    }
  };

  issue_tile(0, 0);
  wait_async_lds();
  __syncthreads();

  for (int s = 0; s < nStages; ++s) {
    const int buf = s & 1;
    if (s + 1 < nStages) issue_tile(s + 1, buf ^ 1);  // overlaps the WMMAs

    // A frag (16x32): lanes 0-15 rows M, K 0..7 & 16..23; lanes 16-31: K 8..15 & 24..31
    // B frag (32x16): lanes 0-15 cols N, K 0..7 & 16..23; lanes 16-31: K 8..15 & 24..31
    Frag fa[2], fb[4];
    const int khA = (lane >> 4) * 8;
    const int khB = (lane >> 4) * 8;
#pragma unroll
    for (int mi = 0; mi < 2; ++mi) {
      int row = waveM * 32 + mi * 16 + (lane & 15);
      fa[mi].q[0] = *(const V4*)&As[buf][row * LDT + khA];
      fa[mi].q[1] = *(const V4*)&As[buf][row * LDT + 16 + khA];
    }
#pragma unroll
    for (int ni = 0; ni < 4; ++ni) {
      int col = waveN * 64 + ni * 16 + (lane & 15);
      fb[ni].q[0] = *(const V4*)&Bs[buf][col * LDT + khB];
      fb[ni].q[1] = *(const V4*)&Bs[buf][col * LDT + 16 + khB];
    }
#pragma unroll
    for (int mi = 0; mi < 2; ++mi)
#pragma unroll
      for (int ni = 0; ni < 4; ++ni)
        acc[mi][ni] = __builtin_amdgcn_wmma_f32_16x16x32_bf16(
            false, fa[mi].v, false, fb[ni].v, (short)0, acc[mi][ni],
            false, false);

    if (s + 1 < nStages) wait_async_lds();  // own copies into buf^1 done
    __syncthreads();                        // everyone done -> LDS consistent
  }

  // epilogue: C/D layout: lane -> col (lane&15), VGPR r -> row r + 8*(lane>>4)
  const int bBatch = (int)(rowBase >> SQ_SHIFT);  // S = 4096 rows per batch
#pragma unroll
  for (int mi = 0; mi < 2; ++mi) {
#pragma unroll
    for (int ni = 0; ni < 4; ++ni) {
      int n = (int)colBase + waveN * 64 + ni * 16 + (lane & 15);
      float bv = 0.f;
      if (biasMode == 1)      bv = bias[n];
      else if (biasMode == 2) bv = bias[bBatch * N + n];
#pragma unroll
      for (int r = 0; r < 8; ++r) {
        int m = (int)rowBase + waveM * 32 + mi * 16 + r + 8 * (lane >> 4);
        float vv = acc[mi][ni][r] + bv;
        if (act == 1)      vv = fmaxf(vv, 0.f);
        else if (act == 2) vv = sigmoidf_dev(vv);
        size_t off = (size_t)m * N + n;
        C[off] = vv;
        if (Cbf) Cbf[off] = f2bf(vv);
      }
    }
  }
}

// ---------------------------------------------------------------- launch
extern "C" void kernel_launch(void* const* d_in, const int* in_sizes, int n_in,
                              void* d_out, int out_size, void* d_ws, size_t ws_size,
                              hipStream_t stream) {
  (void)in_sizes; (void)n_in; (void)out_size; (void)ws_size;
  const float* x    = (const float*)d_in[0];
  const float* W_in = (const float*)d_in[1];
  const float* b_in = (const float*)d_in[2];
  const float* tm_k = (const float*)d_in[3];
  const float* tm_v = (const float*)d_in[4];
  const float* tm_r = (const float*)d_in[5];
  const float* Wk   = (const float*)d_in[6];
  const float* Wv   = (const float*)d_in[7];
  const float* Wr   = (const float*)d_in[8];
  const float* Wo   = (const float*)d_in[9];
  const float* ln1w = (const float*)d_in[10];
  const float* ln1b = (const float*)d_in[11];
  const float* ln2w = (const float*)d_in[12];
  const float* ln2b = (const float*)d_in[13];
  float* out = (float*)d_out;

  char* ws = (char*)d_ws;
  size_t off = 0;
  auto alloc = [&](size_t bytes) -> void* {
    off = (off + 255) & ~(size_t)255;
    void* p = ws + off;
    off += bytes;
    return p;
  };

  unsigned short* x_bf   = (unsigned short*)alloc((size_t)MQ * DIN * 2);
  unsigned short* win_bf = (unsigned short*)alloc((size_t)HQ * DIN * 2);
  unsigned short* wk_bf  = (unsigned short*)alloc((size_t)HQ * HQ * 2);
  unsigned short* wv_bf  = (unsigned short*)alloc((size_t)HQ * HQ * 2);
  unsigned short* wr_bf  = (unsigned short*)alloc((size_t)HQ * HQ * 2);
  unsigned short* wo_bf  = (unsigned short*)alloc((size_t)HQ * HQ * 2);
  float*          h_f    = (float*)alloc((size_t)MQ * HQ * 4);
  unsigned short* h_bf   = (unsigned short*)alloc((size_t)MQ * HQ * 2);
  float*          k_f    = (float*)alloc((size_t)MQ * HQ * 4);
  float*          v_f    = (float*)alloc((size_t)MQ * HQ * 4);
  float*          r_f    = (float*)alloc((size_t)MQ * HQ * 4);
  unsigned short* a_bf   = (unsigned short*)alloc((size_t)MQ * HQ * 2);
  float*          o_f    = (float*)alloc((size_t)MQ * HQ * 4);
  float*          mean   = (float*)alloc((size_t)BQ * HQ * 4);
  float*          bk_b   = (float*)alloc((size_t)BQ * HQ * 4);
  float*          bv_b   = (float*)alloc((size_t)BQ * HQ * 4);
  float*          br_b   = (float*)alloc((size_t)BQ * HQ * 4);

  // input projection: h = x @ W_in^T + b_in  (also stash bf16 copy of h)
  cvt_bf16_kernel<<<1024, 256, 0, stream>>>(x, x_bf, MQ * DIN);
  prep_weight_kernel<<<256, 256, 0, stream>>>(W_in, nullptr, win_bf, HQ, DIN);
  dim3 gemmGrid(HQ / BN, MQ / BM);  // (4, 256)
  gemm_bf16_wmma_kernel<<<gemmGrid, 256, 0, stream>>>(
      x_bf, win_bf, h_f, h_bf, b_in, HQ, DIN, /*act*/0, /*bias*/1);

  for (int l = 0; l < LQ; ++l) {
    const float* Wkl = Wk + (size_t)l * HQ * HQ;
    const float* Wvl = Wv + (size_t)l * HQ * HQ;
    const float* Wrl = Wr + (size_t)l * HQ * HQ;
    const float* Wol = Wo + (size_t)l * HQ * HQ;
    const float* tkl = tm_k + (size_t)l * HQ;
    const float* tvl = tm_v + (size_t)l * HQ;
    const float* trl = tm_r + (size_t)l * HQ;

    seq_mean_kernel<<<(BQ * HQ + 255) / 256, 256, 0, stream>>>(h_f, mean);

    prep_weight_kernel<<<512, 256, 0, stream>>>(Wkl, tkl, wk_bf, HQ, HQ);
    prep_weight_kernel<<<512, 256, 0, stream>>>(Wvl, tvl, wv_bf, HQ, HQ);
    prep_weight_kernel<<<512, 256, 0, stream>>>(Wrl, trl, wr_bf, HQ, HQ);
    prep_weight_kernel<<<512, 256, 0, stream>>>(Wol, nullptr, wo_bf, HQ, HQ);

    int biasBlocks = (BQ * HQ * 32 + 255) / 256;
    mix_bias_kernel<<<biasBlocks, 256, 0, stream>>>(Wkl, tkl, mean, bk_b);
    mix_bias_kernel<<<biasBlocks, 256, 0, stream>>>(Wvl, tvl, mean, bv_b);
    mix_bias_kernel<<<biasBlocks, 256, 0, stream>>>(Wrl, trl, mean, br_b);

    gemm_bf16_wmma_kernel<<<gemmGrid, 256, 0, stream>>>(
        h_bf, wk_bf, k_f, nullptr, bk_b, HQ, HQ, /*relu*/1, 2);
    gemm_bf16_wmma_kernel<<<gemmGrid, 256, 0, stream>>>(
        h_bf, wv_bf, v_f, nullptr, bv_b, HQ, HQ, /*relu*/1, 2);
    gemm_bf16_wmma_kernel<<<gemmGrid, 256, 0, stream>>>(
        h_bf, wr_bf, r_f, nullptr, br_b, HQ, HQ, /*sigmoid*/2, 2);

    wkv_scan_kernel<<<(BQ * HQ + 255) / 256, 256, 0, stream>>>(k_f, v_f, r_f, a_bf);

    gemm_bf16_wmma_kernel<<<gemmGrid, 256, 0, stream>>>(
        a_bf, wo_bf, o_f, nullptr, nullptr, HQ, HQ, /*none*/0, 0);

    float* hout = (l == LQ - 1) ? out : h_f;
    resid_ln_kernel<<<MQ, 256, 0, stream>>>(
        h_f, o_f, ln1w + (size_t)l * HQ, ln1b + (size_t)l * HQ,
        ln2w + (size_t)l * HQ, ln2b + (size_t)l * HQ, hout, h_bf);
  }
}
